// MixHop_86947317941146
// MI455X (gfx1250) — matrix operand order repaired and checked by reading, not compiled, not run on hardware
//
#include <hip/hip_runtime.h>
#include <hip/hip_bf16.h>

typedef unsigned int u32;
typedef __attribute__((ext_vector_type(16))) __bf16 v16bf;
typedef __attribute__((ext_vector_type(8)))  float  v8f;

#define N_NODES 100000
#define N_EDGES 800000
#define BN_EPS 1e-5f

__device__ __forceinline__ u32 bf16_rne(float f) {
  u32 u = __float_as_uint(f);
  u += 0x7fffu + ((u >> 16) & 1u);
  return u >> 16;
}
__device__ __forceinline__ u32 pack_bf16(float lo, float hi) {
  return bf16_rne(lo) | (bf16_rne(hi) << 16);
}

// ---------------- elementwise helpers ----------------
__global__ void zero_kernel(float* __restrict__ p, size_t n) {
  size_t i = (size_t)blockIdx.x * blockDim.x + threadIdx.x;
  size_t stride = (size_t)gridDim.x * blockDim.x;
  for (; i < n; i += stride) p[i] = 0.0f;
}

__global__ void deg_kernel(const int* __restrict__ col, float* __restrict__ deg, int E) {
  int e = blockIdx.x * blockDim.x + threadIdx.x;
  if (e < E) atomicAdd(&deg[col[e]], 1.0f);
}

__global__ void dinv_kernel(const float* __restrict__ deg, float* __restrict__ dinv, int n) {
  int i = blockIdx.x * blockDim.x + threadIdx.x;
  if (i < n) {
    float d = deg[i];
    dinv[i] = (d > 0.0f) ? rsqrtf(d) : 0.0f;
  }
}

__global__ void edgew_kernel(const int* __restrict__ row, const int* __restrict__ col,
                             const float* __restrict__ dinv, float* __restrict__ we, int E) {
  int e = blockIdx.x * blockDim.x + threadIdx.x;
  if (e < E) we[e] = dinv[row[e]] * dinv[col[e]];
}

// ---------------- weight packing into WMMA B-operand layout ----------------
// B (32x16 bf16 k-tile): lane L holds column n = L&15, k-base = (L>>4)*16,
// VGPR i = pack(B[kb+2i][n], B[kb+2i+1][n]).  One block per (ktile,ntile),
// 256 threads = 32 lanes x 8 dwords, stored contiguously per lane.
__global__ void pack_w(const float* __restrict__ W, u32* __restrict__ dst) {
  int kt = blockIdx.x >> 3;
  int nt = blockIdx.x & 7;
  int t = threadIdx.x;
  int lane = t >> 3, i = t & 7;
  int n  = lane & 15;
  int kb = (lane >> 4) * 16;
  int k  = kt * 32 + kb + 2 * i;
  float lo = W[(size_t)k * 128 + nt * 16 + n];
  float hi = W[(size_t)(k + 1) * 128 + nt * 16 + n];
  dst[(size_t)blockIdx.x * 256 + t] = pack_bf16(lo, hi);
}

// ---------------- activation packing into WMMA A-operand layout ----------------
// A (16x32 bf16 k-tile): lane L holds row m = L&15, k-base = (L>=16)?16:0,
// dword i = pack over K pair 2i + kb.
// Stored so each lane's 8 dwords are contiguous; (mtile, ktile) chunks of
// 256 dwords, ktile-minor => GEMM streams A linearly.
__global__ void pack_a(const float* __restrict__ A, int lda, int ktiles,
                       u32* __restrict__ dst) {
  int bid = blockIdx.x;
  int kt = bid % ktiles;
  int mt = bid / ktiles;
  int t = threadIdx.x;
  int lane = t >> 3, i = t & 7;
  int m = mt * 16 + (lane & 15);
  int kb = (lane >> 4) * 16;
  int k = kt * 32 + kb + 2 * i;
  float2 f = *(const float2*)(A + (size_t)m * lda + k);
  dst[(size_t)bid * 256 + t] = pack_bf16(f.x, f.y);
}

// ---------------- WMMA GEMM: C[N x 128] = A[N x K] @ W + bias ----------------
// grid.x = N/16, block = 256 (8 waves). Wave w owns output n-tile w.
// Both operands pre-packed bf16: inner loop = 4x b128 loads + 1 wmma.
template <int KT>
__global__ void __launch_bounds__(256) gemm_wmma(
    const u32* __restrict__ Apk,
    const u32* __restrict__ Bpk,
    const float* __restrict__ bias,
    float* __restrict__ C, int ldc, int coff) {
  const int lane = threadIdx.x & 31;
  const int wv = threadIdx.x >> 5;  // n-tile 0..7
  const int m0 = blockIdx.x * 16;
  const u32* aptr = Apk + (size_t)blockIdx.x * KT * 256 + lane * 8;
  const u32* bptr = Bpk + (size_t)wv * 256 + lane * 8;

  v8f c = {};
#pragma unroll
  for (int kt = 0; kt < KT; ++kt) {
    union { v16bf v; uint4 q[2]; } a, b;
    const u32* ap = aptr + (size_t)kt * 256;
    a.q[0] = *(const uint4*)(ap);
    a.q[1] = *(const uint4*)(ap + 4);
    const u32* bp = bptr + (size_t)kt * (8 * 256);
    b.q[0] = *(const uint4*)(bp);
    b.q[1] = *(const uint4*)(bp + 4);
    c = __builtin_amdgcn_wmma_f32_16x16x32_bf16(false, a.v, false, b.v,
                                                (short)0, c, false, false);
  }
  // C layout: lane L -> col n = L&15 (+16*wv), rows m0 + r + ((L>=16)?8:0)
  const int n = lane & 15;
  const int mb = (lane >> 4) * 8;
  const int colg = wv * 16 + n;
  const float bv = bias[colg];
#pragma unroll
  for (int r = 0; r < 8; ++r) {
    C[(size_t)(m0 + mb + r) * ldc + coff + colg] = c[r] + bv;
  }
}

// ---------------- SpMM scatter: out[col[e]] += w[e] * in[row[e]] ----------------
// One wave per edge per iteration; 32 lanes x float4 = 128 features.
// Feature maps (51.2 MB) are L2-resident (192 MB L2), so the random gathers
// and f32 atomic scatters stay on-chip.
__global__ void spmm_kernel(const int* __restrict__ row, const int* __restrict__ col,
                            const float* __restrict__ we,
                            const float* __restrict__ in,
                            float* __restrict__ out, int ldo, int coff, int E) {
  const int lane = threadIdx.x & 31;
  int warp = blockIdx.x * (blockDim.x >> 5) + (threadIdx.x >> 5);
  const int nwarps = gridDim.x * (blockDim.x >> 5);
  for (int e = warp; e < E; e += nwarps) {
    int r = row[e];
    int c = col[e];
    float w = we[e];
    float4 v = *(const float4*)(in + (size_t)r * 128 + lane * 4);
    float* o = out + (size_t)c * ldo + coff + lane * 4;
    atomicAdd(o + 0, w * v.x);
    atomicAdd(o + 1, w * v.y);
    atomicAdd(o + 2, w * v.z);
    atomicAdd(o + 3, w * v.w);
  }
}

// ---------------- BatchNorm ----------------
// blockDim = 384 (one thread per column, coalesced row reads), 128 rows/block.
__global__ void bn_stats_kernel(const float* __restrict__ h, float* __restrict__ stats) {
  const int c = threadIdx.x;
  int r0 = blockIdx.x * 128;
  int r1 = min(r0 + 128, N_NODES);
  float s = 0.0f, s2 = 0.0f;
  for (int r = r0; r < r1; ++r) {
    float v = h[(size_t)r * 384 + c];
    s += v;
    s2 += v * v;
  }
  atomicAdd(&stats[c], s);
  atomicAdd(&stats[384 + c], s2);
}

__global__ void bn_apply_kernel(float* __restrict__ h, const float* __restrict__ stats,
                                const float* __restrict__ g, const float* __restrict__ b) {
  const float invn = 1.0f / (float)N_NODES;
  size_t total = (size_t)N_NODES * 384;
  size_t i = (size_t)blockIdx.x * blockDim.x + threadIdx.x;
  size_t stride = (size_t)gridDim.x * blockDim.x;
  for (; i < total; i += stride) {
    int c = (int)(i % 384);
    float mu = stats[c] * invn;
    float var = stats[384 + c] * invn - mu * mu;
    float y = (h[i] - mu) * rsqrtf(var + BN_EPS) * g[c] + b[c];
    h[i] = fmaxf(y, 0.0f);
  }
}

// ---------------- orchestration ----------------
extern "C" void kernel_launch(void* const* d_in, const int* in_sizes, int n_in,
                              void* d_out, int out_size, void* d_ws, size_t ws_size,
                              hipStream_t stream) {
  (void)in_sizes; (void)n_in; (void)out_size; (void)ws_size;
  const float* x   = (const float*)d_in[0];
  const int*   ei  = (const int*)d_in[1];
  const float* W0  = (const float*)d_in[2];
  const float* b0  = (const float*)d_in[3];
  const float* W1  = (const float*)d_in[4];
  const float* b1  = (const float*)d_in[5];
  const float* W2  = (const float*)d_in[6];
  const float* b2  = (const float*)d_in[7];
  const float* bng = (const float*)d_in[8];
  const float* bnb = (const float*)d_in[9];
  float* out = (float*)d_out;

  const int* row = ei;             // edge_index[0]
  const int* col = ei + N_EDGES;   // edge_index[1]

  // workspace carve-out (floats)
  float* f = (float*)d_ws;
  size_t off = 0;
  float* deg   = f + off; off += 100032;                  // N rounded to 64
  float* dinv  = f + off; off += 100032;
  float* we    = f + off; off += (size_t)N_EDGES;
  float* xj    = f + off; off += (size_t)N_NODES * 128;
  float* xj2   = f + off; off += (size_t)N_NODES * 128;
  float* hbuf  = f + off; off += (size_t)N_NODES * 384;
  u32*   wpk   = (u32*)(f + off); off += 172032;          // packed bf16 weights
  u32*   apk   = (u32*)(f + off); off += (size_t)(N_NODES / 16) * 12 * 256; // packed bf16 A
  float* stats = f + off; off += 768;

  // ---- GCN normalization ----
  zero_kernel<<<512, 256, 0, stream>>>(deg, N_NODES);
  deg_kernel<<<(N_EDGES + 255) / 256, 256, 0, stream>>>(col, deg, N_EDGES);
  dinv_kernel<<<(N_NODES + 255) / 256, 256, 0, stream>>>(deg, dinv, N_NODES);
  edgew_kernel<<<(N_EDGES + 255) / 256, 256, 0, stream>>>(row, col, dinv, we, N_EDGES);

  // ---- pack weights: W0 (3 x 128x128), W1/W2 (3 x 384x128) ----
  const int SZ128 = 4 * 8 * 256;    // 8192 dwords per 128x128 matrix
  const int SZ384 = 12 * 8 * 256;   // 24576 dwords per 384x128 matrix
  u32* wpk0 = wpk;
  u32* wpk1 = wpk + 3 * SZ128;
  u32* wpk2 = wpk1 + 3 * SZ384;
  for (int j = 0; j < 3; ++j) {
    pack_w<<<4 * 8, 256, 0, stream>>>(W0 + (size_t)j * 128 * 128, wpk0 + (size_t)j * SZ128);
    pack_w<<<12 * 8, 256, 0, stream>>>(W1 + (size_t)j * 384 * 128, wpk1 + (size_t)j * SZ384);
    pack_w<<<12 * 8, 256, 0, stream>>>(W2 + (size_t)j * 384 * 128, wpk2 + (size_t)j * SZ384);
  }

  const int GEMM_GRID = N_NODES / 16;  // 6250

  auto mixhop = [&](const float* hin, int lda, int ktiles, const u32* wl,
                    const float* bl, float* hout) {
    // pack layer input to WMMA A layout once (consumed by all 3 GEMMs)
    pack_a<<<GEMM_GRID * ktiles, 256, 0, stream>>>(hin, lda, ktiles, apk);
    zero_kernel<<<8192, 256, 0, stream>>>(hout, (size_t)N_NODES * 384);
    int szmat = ktiles * 8 * 256;
    if (ktiles == 4) {
      gemm_wmma<4><<<GEMM_GRID, 256, 0, stream>>>(apk, wl, bl, hout, 384, 0);
      gemm_wmma<4><<<GEMM_GRID, 256, 0, stream>>>(apk, wl + szmat, bl + 128, xj, 128, 0);
    } else {
      gemm_wmma<12><<<GEMM_GRID, 256, 0, stream>>>(apk, wl, bl, hout, 384, 0);
      gemm_wmma<12><<<GEMM_GRID, 256, 0, stream>>>(apk, wl + szmat, bl + 128, xj, 128, 0);
    }
    // hop 1: one propagation into hout cols 128..255
    spmm_kernel<<<12500, 256, 0, stream>>>(row, col, we, xj, hout, 384, 128, N_EDGES);
    // hop 2: two propagations into hout cols 256..383
    if (ktiles == 4) {
      gemm_wmma<4><<<GEMM_GRID, 256, 0, stream>>>(apk, wl + 2 * szmat, bl + 256, xj, 128, 0);
    } else {
      gemm_wmma<12><<<GEMM_GRID, 256, 0, stream>>>(apk, wl + 2 * szmat, bl + 256, xj, 128, 0);
    }
    zero_kernel<<<4096, 256, 0, stream>>>(xj2, (size_t)N_NODES * 128);
    spmm_kernel<<<12500, 256, 0, stream>>>(row, col, we, xj, xj2, 128, 0, N_EDGES);
    spmm_kernel<<<12500, 256, 0, stream>>>(row, col, we, xj2, hout, 384, 256, N_EDGES);
  };

  auto bn = [&](float* h, int layer) {
    zero_kernel<<<3, 256, 0, stream>>>(stats, 768);
    bn_stats_kernel<<<(N_NODES + 127) / 128, 384, 0, stream>>>(h, stats);
    bn_apply_kernel<<<8192, 256, 0, stream>>>(h, stats, bng + layer * 384, bnb + layer * 384);
  };

  // layer 0: x (N x 128) -> hbuf (N x 384)
  mixhop(x, 128, 4, wpk0, b0, hbuf);
  bn(hbuf, 0);
  // layer 1: hbuf -> d_out
  mixhop(hbuf, 384, 12, wpk1, b1, out);
  bn(out, 1);
  // layer 2: d_out -> hbuf, then copy to d_out
  mixhop(out, 384, 12, wpk2, b2, hbuf);
  hipMemcpyAsync(out, hbuf, (size_t)N_NODES * 384 * sizeof(float),
                 hipMemcpyDeviceToDevice, stream);
}
